// PIFormer_21131239097226
// MI455X (gfx1250) — matrix operand order repaired and checked by reading, not compiled, not run on hardware
//
#include <hip/hip_runtime.h>
#include <hip/hip_bf16.h>
#include <math.h>
#include <stdint.h>

// ---------------- types / constants ----------------
typedef _Float16 half_t;
typedef __attribute__((ext_vector_type(16))) _Float16 v16h;
typedef __attribute__((ext_vector_type(8)))  _Float16 v8h;
typedef __attribute__((ext_vector_type(4)))  _Float16 v4h;
typedef __attribute__((ext_vector_type(8)))  float    v8f;

static constexpr int HEADS    = 8;
static constexpr int HEAD_DIM = 32;
static constexpr int NODE_CH  = 512;
static constexpr int PAIR_CH  = 128;
static constexpr int MAX_PI   = 4;
static constexpr int N_STACKS = 4;
static constexpr int CNU      = 5000;
static constexpr int CNUV     = 40000;
static constexpr int CNIJK    = 10000;
static constexpr int CNIJKL   = 40000;
static constexpr int CNUIJK   = 80000;
static constexpr int CNUIJKL  = 320000;
static constexpr int CNI      = 5000;
static constexpr int CNUI     = 20000;
static constexpr int NNODE    = CNU + CNIJK;                    // 15000
static constexpr int NEDGE    = CNUV + 2*CNUIJK + CNIJKL;       // 240000
static constexpr int QKDIM    = HEADS*HEAD_DIM;                 // 256

#define CDIV(a,b) (((a)+(b)-1)/(b))

// ---------------- device helpers ----------------
__device__ __forceinline__ float gelu_f(float x) {
  const float c = 0.7978845608028654f;
  return 0.5f * x * (1.0f + tanhf(c * (x + 0.044715f * x * x * x)));
}
__device__ __forceinline__ unsigned fenc(float f) {
  unsigned u = __float_as_uint(f);
  return (u & 0x80000000u) ? ~u : (u | 0x80000000u);
}
__device__ __forceinline__ float fdec(unsigned u) {
  unsigned b = (u & 0x80000000u) ? (u & 0x7FFFFFFFu) : ~u;
  return __uint_as_float(b);
}
static constexpr unsigned NEG_INF_ENC = 0x007FFFFFu; // fenc(-inf)

// ---------------- generic fills / copies ----------------
__global__ void k_fill_f32(float* __restrict__ p, float v, long n) {
  long t = (long)blockIdx.x * blockDim.x + threadIdx.x;
  if (t < n) p[t] = v;
}
__global__ void k_fill_u32(unsigned* __restrict__ p, unsigned v, long n) {
  long t = (long)blockIdx.x * blockDim.x + threadIdx.x;
  if (t < n) p[t] = v;
}
__global__ void k_copy_f32x4(float4* __restrict__ d, const float4* __restrict__ s, long n4) {
  long t = (long)blockIdx.x * blockDim.x + threadIdx.x;
  if (t < n4) d[t] = s[t];
}
__global__ void k_gelu_arr(float* __restrict__ d, const float* __restrict__ s, long n) {
  long t = (long)blockIdx.x * blockDim.x + threadIdx.x;
  if (t < n) d[t] = gelu_f(s[t]);
}
__global__ void k_prop_fin(float* __restrict__ p, const float* __restrict__ t_, long n) {
  long t = (long)blockIdx.x * blockDim.x + threadIdx.x;
  if (t < n) p[t] = 2.0f * p[t] + t_[t];
}
__global__ void k_gather_rows4(float4* __restrict__ dst, const float4* __restrict__ src,
                               const int* __restrict__ idx, int rows, int cols4, int rowOff) {
  long t = (long)blockIdx.x * blockDim.x + threadIdx.x;
  if (t >= (long)rows * cols4) return;
  int r = (int)(t / cols4), c = (int)(t % cols4);
  dst[t] = src[(long)(idx[r] + rowOff) * cols4 + c];
}

// ---------------- WMMA GEMM path (fully specialized) ----------------
// B fragment: lane l holds col n=l&15; element e -> k = e + (l>=16 ? 16 : 0)
__global__ void k_pack_b(const float* __restrict__ B, half_t* __restrict__ Bp,
                         int K, int N, int KT, int NT) {
  long total = (long)NT * KT * 512;
  long t = (long)blockIdx.x * blockDim.x + threadIdx.x;
  if (t >= total) return;
  int e    = (int)(t & 15);
  int lane = (int)((t >> 4) & 31);
  long f   = t >> 9;               // fragment id = nt*KT + kt
  int kt   = (int)(f % KT);
  int nt   = (int)(f / KT);
  int n = nt * 16 + (lane & 15);
  int k = kt * 32 + e + ((lane >= 16) ? 16 : 0);
  Bp[t] = (k < K && n < N) ? (half_t)B[(long)k * N + n] : (half_t)0.0f;
}

static constexpr int ARS = 40;  // LDS A-tile row stride in halfs (80B: 16B aligned)

// Stage the 16x32 f32 A tile (m0, k0) into LDS as f16 (row-major, padded stride).
// GK: guard k range (tail tile / K%4!=0). GM: guard m range (edge M tile).
template<int K, int NTH, bool GK, bool GM>
__device__ __forceinline__ void stage_a(const float* __restrict__ A, _Float16* __restrict__ aS,
                                        int m0, int k0, int M) {
#pragma unroll
  for (int i = threadIdx.x; i < 128; i += NTH) {
    const int r  = i >> 3;
    const int c4 = (i & 7) << 2;
    const int m  = m0 + r;
    v4h h4; h4[0] = (half_t)0.f; h4[1] = (half_t)0.f; h4[2] = (half_t)0.f; h4[3] = (half_t)0.f;
    if (!GM || m < M) {
      if (!GK) {
        const float4 f = *reinterpret_cast<const float4*>(A + (long)m * K + (k0 + c4));
        h4[0] = (half_t)f.x; h4[1] = (half_t)f.y; h4[2] = (half_t)f.z; h4[3] = (half_t)f.w;
      } else {
#pragma unroll
        for (int j = 0; j < 4; ++j) {
          const int k = k0 + c4 + j;
          if (k < K) h4[j] = (half_t)A[(long)m * K + k];
        }
      }
    }
    *reinterpret_cast<v4h*>(&aS[r * ARS + c4]) = h4;
  }
}

__device__ __forceinline__ v8f frag_wmma(const _Float16* __restrict__ aS,
                                         const half_t* __restrict__ bp,
                                         v8f acc, int lane) {
  const int r   = lane & 15;
  const int off = (lane >= 16) ? 16 : 0;
  const v8h lo = *reinterpret_cast<const v8h*>(&aS[r * ARS + off]);
  const v8h hi = *reinterpret_cast<const v8h*>(&aS[r * ARS + off + 8]);
  v16h af;
#pragma unroll
  for (int j = 0; j < 8; ++j) { af[j] = lo[j]; af[j + 8] = hi[j]; }
  const v16h bf = *reinterpret_cast<const v16h*>(bp);
  return __builtin_amdgcn_wmma_f32_16x16x32_f16(false, af, false, bf,
                                                (short)0, acc, false, false);
}

// C[M, N (ldc=LDC)] = ACT(A[M,K] x Bp + bias) (+C if ADDC).
// blockDim = 32*W, W = min(NT,8); grid = (M/16, NT/W) -- NT is always divisible by W.
template<int K, int N, int LDC, int ACT, int ADDC>
__global__ __launch_bounds__((((N + 15) / 16) < 8 ? ((N + 15) / 16) : 8) * 32)
void k_gemm_t(const float* __restrict__ A, const half_t* __restrict__ Bp,
              const float* __restrict__ bias, float* __restrict__ C, int M) {
  constexpr int KT    = (K + 31) / 32;
  constexpr int NT    = (N + 15) / 16;
  constexpr int W     = NT < 8 ? NT : 8;
  constexpr int NTH   = 32 * W;
  constexpr int KFULL = ((K % 4) == 0) ? (K / 32) : 0;  // tiles needing no k-guard
  __shared__ __align__(16) _Float16 aS[2][16 * ARS];
  const int lane = threadIdx.x & 31;
  const int wave = threadIdx.x >> 5;
  const int m0   = blockIdx.x * 16;
  const int nt   = blockIdx.y * W + wave;
  const bool fullM = (m0 + 16 <= M);
  const half_t* bp = Bp + ((long)nt * KT * 32 + lane) * 16;
  v8f acc = {};
  // stage tile 0
  if (fullM) {
    if (KFULL > 0) stage_a<K, NTH, false, false>(A, aS[0], m0, 0, M);
    else           stage_a<K, NTH, true,  false>(A, aS[0], m0, 0, M);
  } else {
    if (KFULL > 0) stage_a<K, NTH, false, true >(A, aS[0], m0, 0, M);
    else           stage_a<K, NTH, true,  true >(A, aS[0], m0, 0, M);
  }
  __syncthreads();
  for (int kt = 0; kt < KT; ++kt) {
    if (kt + 1 < KT) {           // double-buffer: stage next while computing current
      const int k0 = (kt + 1) * 32;
      const bool gk = (kt + 1 >= KFULL);
      if (fullM) {
        if (gk) stage_a<K, NTH, true,  false>(A, aS[(kt + 1) & 1], m0, k0, M);
        else    stage_a<K, NTH, false, false>(A, aS[(kt + 1) & 1], m0, k0, M);
      } else {
        if (gk) stage_a<K, NTH, true,  true >(A, aS[(kt + 1) & 1], m0, k0, M);
        else    stage_a<K, NTH, false, true >(A, aS[(kt + 1) & 1], m0, k0, M);
      }
      __builtin_prefetch(bp + (long)(kt + 1) * 512, 0, 3);  // next B fragment
    }
    acc = frag_wmma(aS[kt & 1], bp + (long)kt * 512, acc, lane);
    __syncthreads();
  }
  const int n = nt * 16 + (lane & 15);
  if (N % 16 != 0 && n >= N) return;
  const int mb = (lane >= 16) ? 8 : 0;
  const float bv = bias ? bias[n] : 0.0f;
  if (fullM) {
#pragma unroll
    for (int r = 0; r < 8; ++r) {
      float v = acc[r] + bv;
      if (ACT == 1) v = gelu_f(v);
      const long o = (long)(m0 + mb + r) * LDC + n;
      if (ADDC) v += C[o];
      C[o] = v;
    }
  } else {
#pragma unroll
    for (int r = 0; r < 8; ++r) {
      const int m = m0 + mb + r;
      if (m >= M) continue;
      float v = acc[r] + bv;
      if (ACT == 1) v = gelu_f(v);
      const long o = (long)m * LDC + n;
      if (ADDC) v += C[o];
      C[o] = v;
    }
  }
}

template<int K, int N, int LDC, int ACT, int ADDC>
static inline void run_gemm_t(hipStream_t st, const float* A, const float* B, const float* bias,
                              float* C, int M, half_t* packb) {
  constexpr int KT = (K + 31) / 32, NT = (N + 15) / 16;
  constexpr int W  = NT < 8 ? NT : 8;
  long pe = (long)NT * KT * 512;
  k_pack_b<<<(unsigned)CDIV(pe, 256), 256, 0, st>>>(B, packb, K, N, KT, NT);
  dim3 grid(CDIV(M, 16), NT / W);
  k_gemm_t<K, N, LDC, ACT, ADDC><<<grid, 32 * W, 0, st>>>(A, packb, bias, C, M);
}

// ---------------- LayerNorm (one row per wave32) ----------------
__global__ void k_layernorm(const float* __restrict__ x, const float* __restrict__ s,
                            const float* __restrict__ b, float* __restrict__ y,
                            int rows, int C) {
  int row = blockIdx.x * blockDim.y + threadIdx.y;
  if (row >= rows) return;
  int lane = threadIdx.x;
  const float4* xr = reinterpret_cast<const float4*>(x + (long)row * C);
  const int C4 = C >> 2;
  float sum = 0.f, sq = 0.f;
  for (int j = lane; j < C4; j += 32) {
    float4 v = xr[j];
    sum += v.x + v.y + v.z + v.w;
    sq  += v.x * v.x + v.y * v.y + v.z * v.z + v.w * v.w;
  }
#pragma unroll
  for (int off = 16; off > 0; off >>= 1) {
    sum += __shfl_xor(sum, off);
    sq  += __shfl_xor(sq,  off);
  }
  float mu  = sum / C;
  float var = sq / C - mu * mu;
  float inv = rsqrtf(var + 1e-5f);
  float4* yr = reinterpret_cast<float4*>(y + (long)row * C);
  const float4* sv = reinterpret_cast<const float4*>(s);
  const float4* bv = reinterpret_cast<const float4*>(b);
  for (int j = lane; j < C4; j += 32) {
    float4 v = xr[j], ss = sv[j], bb = bv[j], o;
    o.x = (v.x - mu) * inv * ss.x + bb.x;
    o.y = (v.y - mu) * inv * ss.y + bb.y;
    o.z = (v.z - mu) * inv * ss.z + bb.z;
    o.w = (v.w - mu) * inv * ss.w + bb.w;
    yr[j] = o;
  }
}

// ---------------- graph / attention kernels ----------------
__global__ void k_build_sd(int* __restrict__ src, int* __restrict__ dst,
                           const int* __restrict__ uv_u, const int* __restrict__ uv_v,
                           const int* __restrict__ Uijk_u, const int* __restrict__ Uijk_ijk,
                           const int* __restrict__ ijkl_ijk, const int* __restrict__ ijkl_jkl) {
  int t = blockIdx.x * blockDim.x + threadIdx.x;
  if (t >= NEDGE) return;
  if (t < CNUV)                      { src[t] = uv_u[t];                 dst[t] = uv_v[t]; }
  else if (t < CNUV + CNUIJK)        { int i = t - CNUV;
                                       src[t] = Uijk_u[i];               dst[t] = CNU + Uijk_ijk[i]; }
  else if (t < CNUV + 2 * CNUIJK)    { int i = t - CNUV - CNUIJK;
                                       src[t] = CNU + Uijk_ijk[i];       dst[t] = Uijk_u[i]; }
  else                               { int i = t - CNUV - 2 * CNUIJK;
                                       src[t] = CNU + ijkl_ijk[i];       dst[t] = CNU + ijkl_jkl[i]; }
}

__global__ void k_logits(const float* __restrict__ q, const float* __restrict__ k,
                         const int* __restrict__ src, const int* __restrict__ dst,
                         const float* __restrict__ ebias, float* __restrict__ logits,
                         int ne) {
  long t = (long)blockIdx.x * blockDim.x + threadIdx.x;
  if (t >= (long)ne * HEADS) return;
  int e = (int)(t >> 3), h = (int)(t & 7);
  const float4* qp = reinterpret_cast<const float4*>(q + (long)dst[e] * QKDIM + h * HEAD_DIM);
  const float4* kp = reinterpret_cast<const float4*>(k + (long)src[e] * QKDIM + h * HEAD_DIM);
  float acc = 0.f;
#pragma unroll
  for (int j = 0; j < HEAD_DIM / 4; ++j) {
    float4 a = qp[j], b = kp[j];
    acc += a.x * b.x + a.y * b.y + a.z * b.z + a.w * b.w;
  }
  logits[t] = acc * 0.17677669529663687f + ebias[t];   // 1/sqrt(32)
}

__global__ void k_seg_max(const float* __restrict__ x, const int* __restrict__ seg,
                          unsigned* __restrict__ smax, int rows) {
  long t = (long)blockIdx.x * blockDim.x + threadIdx.x;
  if (t >= (long)rows * HEADS) return;
  int r = (int)(t >> 3), h = (int)(t & 7);
  atomicMax(&smax[(long)seg[r] * HEADS + h], fenc(x[t]));
}
__global__ void k_seg_expsum(const float* __restrict__ x, const int* __restrict__ seg,
                             const unsigned* __restrict__ smax, float* __restrict__ ex,
                             float* __restrict__ ssum, int rows) {
  long t = (long)blockIdx.x * blockDim.x + threadIdx.x;
  if (t >= (long)rows * HEADS) return;
  int r = (int)(t >> 3), h = (int)(t & 7);
  float m = fdec(smax[(long)seg[r] * HEADS + h]);
  if (!isfinite(m)) m = 0.0f;
  float e = expf(x[t] - m);
  ex[t] = e;
  atomicAdd(&ssum[(long)seg[r] * HEADS + h], e);
}
__global__ void k_seg_norm(const float* __restrict__ ex, const float* __restrict__ ssum,
                           const int* __restrict__ seg, float* __restrict__ alpha, int rows) {
  long t = (long)blockIdx.x * blockDim.x + threadIdx.x;
  if (t >= (long)rows * HEADS) return;
  int r = (int)(t >> 3), h = (int)(t & 7);
  alpha[t] = ex[t] / (ssum[(long)seg[r] * HEADS + h] + 1e-9f);
}

__global__ void k_attn_msg(const float* __restrict__ ex, const float* __restrict__ ssum,
                           const float* __restrict__ v, const int* __restrict__ src,
                           const int* __restrict__ dst, float* __restrict__ msg, int ne) {
  long t = (long)blockIdx.x * blockDim.x + threadIdx.x;
  if (t >= (long)ne * HEADS) return;
  int e = (int)(t >> 3), h = (int)(t & 7);
  int d = dst[e];
  float alpha = ex[t] / (ssum[(long)d * HEADS + h] + 1e-9f);
  const float4* vp = reinterpret_cast<const float4*>(v + (long)src[e] * QKDIM + h * HEAD_DIM);
  float* mp = msg + (long)d * QKDIM + h * HEAD_DIM;
#pragma unroll
  for (int j = 0; j < HEAD_DIM / 4; ++j) {
    float4 vv = vp[j];
    atomicAdd(&mp[j * 4 + 0], alpha * vv.x);
    atomicAdd(&mp[j * 4 + 1], alpha * vv.y);
    atomicAdd(&mp[j * 4 + 2], alpha * vv.z);
    atomicAdd(&mp[j * 4 + 3], alpha * vv.w);
  }
}

// outs buffer NUIJK x 40 (5 slots of 8); iteration `it` reads slot it, scatters slot it+1
__global__ void k_path_iter(float* __restrict__ outs, const float* __restrict__ alpha,
                            const int* __restrict__ i_ijkl, const int* __restrict__ i_Uijk,
                            const int* __restrict__ i_ujkl, int it) {
  long t = (long)blockIdx.x * blockDim.x + threadIdx.x;
  if (t >= (long)CNUIJKL * HEADS) return;
  int e = (int)(t >> 3), h = (int)(t & 7);
  float a = alpha[(long)i_ijkl[e] * HEADS + h];
  float g = outs[(long)i_Uijk[e] * 40 + it * 8 + h];
  atomicAdd(&outs[(long)i_ujkl[e] * 40 + (it + 1) * 8 + h], a * g);
}
__global__ void k_zero_path_cols(float* __restrict__ outs) {
  long t = (long)blockIdx.x * blockDim.x + threadIdx.x;
  if (t >= (long)CNUIJK * 32) return;
  int r = (int)(t / 32), c = 8 + (int)(t % 32);
  outs[(long)r * 40 + c] = 0.0f;
}

// ---------------- host-side helpers ----------------
static inline void run_ln(hipStream_t st, const float* x, const float* s, const float* b,
                          float* y, int rows, int C) {
  dim3 blk(32, 8);
  k_layernorm<<<CDIV(rows, 8), blk, 0, st>>>(x, s, b, y, rows, C);
}
static inline void d2d4(hipStream_t st, float* d, const float* s, long n) {
  k_copy_f32x4<<<(unsigned)CDIV(n / 4, 256), 256, 0, st>>>(
      reinterpret_cast<float4*>(d), reinterpret_cast<const float4*>(s), n / 4);
}

struct StackP {
  const float *pmlp_b1, *pmlp_b2, *pmlp_w1, *pmlp_w2, *w_tri, *w_val;
  const float *b_eu, *bo, *ln1_b, *ln1_s, *ln2_b, *ln2_s;
  const float *mlp_b1, *mlp_b2, *mlp_w1, *mlp_w2, *w_eu, *wb, *wk, *wo, *wq, *wv;
};

extern "C" void kernel_launch(void* const* d_in, const int* in_sizes, int n_in,
                              void* d_out, int out_size, void* d_ws, size_t ws_size,
                              hipStream_t stream) {
  (void)in_sizes; (void)n_in; (void)out_size; (void)ws_size;
  // ---- positional inputs (setup_inputs insertion order) ----
  const float* ae   = (const float*)d_in[0];   // [NU,23]
  const float* ape  = (const float*)d_in[1];   // [NUV,8]
  const float* afpe = (const float*)d_in[2];   // [NUIJK,24]
  const float* se   = (const float*)d_in[3];   // [NIJKL,14]
  const int* uv_u       = (const int*)d_in[4];
  const int* uv_v       = (const int*)d_in[5];
  const int* Uijk_u     = (const int*)d_in[6];
  const int* Uijk_ijk   = (const int*)d_in[7];
  const int* ijkl_ijk   = (const int*)d_in[8];
  const int* ijkl_jkl   = (const int*)d_in[9];
  const int* Uijkl_ijkl = (const int*)d_in[10];
  const int* Uijkl_Uijk = (const int*)d_in[11];
  const int* Uijkl_ujkl = (const int*)d_in[12];
  const int* I_ijk      = (const int*)d_in[13];
  const int* UI_Uijk    = (const int*)d_in[14];
  // d_in[15] = num_ijk scalar (compile-time CNIJK here)
  // ---- params: jax pytree leaves (sorted dict keys, list order) from 16 ----
  int p = 16;
  const float* in_atom_b   = (const float*)d_in[p++];
  const float* in_atom_w   = (const float*)d_in[p++];
  const float* in_pair_b   = (const float*)d_in[p++];
  const float* in_pair_w   = (const float*)d_in[p++];
  const float* in_prop_b1  = (const float*)d_in[p++];
  const float* in_prop_b2  = (const float*)d_in[p++];
  const float* in_prop_w1  = (const float*)d_in[p++];
  const float* in_prop_w2  = (const float*)d_in[p++];
  const float* in_stereo_b = (const float*)d_in[p++];
  const float* in_stereo_w = (const float*)d_in[p++];
  StackP st[N_STACKS];
  for (int s = 0; s < N_STACKS; ++s) {
    st[s].pmlp_b1 = (const float*)d_in[p++]; st[s].pmlp_b2 = (const float*)d_in[p++];
    st[s].pmlp_w1 = (const float*)d_in[p++]; st[s].pmlp_w2 = (const float*)d_in[p++];
    st[s].w_tri   = (const float*)d_in[p++]; st[s].w_val   = (const float*)d_in[p++];
    st[s].b_eu    = (const float*)d_in[p++]; st[s].bo      = (const float*)d_in[p++];
    st[s].ln1_b   = (const float*)d_in[p++]; st[s].ln1_s   = (const float*)d_in[p++];
    st[s].ln2_b   = (const float*)d_in[p++]; st[s].ln2_s   = (const float*)d_in[p++];
    st[s].mlp_b1  = (const float*)d_in[p++]; st[s].mlp_b2  = (const float*)d_in[p++];
    st[s].mlp_w1  = (const float*)d_in[p++]; st[s].mlp_w2  = (const float*)d_in[p++];
    st[s].w_eu    = (const float*)d_in[p++]; st[s].wb      = (const float*)d_in[p++];
    st[s].wk      = (const float*)d_in[p++]; st[s].wo      = (const float*)d_in[p++];
    st[s].wq      = (const float*)d_in[p++]; st[s].wv      = (const float*)d_in[p++];
  }
  // ---- workspace carve (~390 MB) ----
  uint8_t* wsb = (uint8_t*)d_ws; size_t off = 0;
  auto wsAlloc = [&](size_t bytes) -> void* {
    off = (off + 255) & ~(size_t)255; void* q = wsb + off; off += bytes; return q;
  };
  float* nodes  = (float*)wsAlloc((size_t)NNODE * NODE_CH * 4);
  float* hbuf   = (float*)wsAlloc((size_t)NNODE * NODE_CH * 4);
  float* mlpt   = (float*)wsAlloc((size_t)NNODE * NODE_CH * 4);
  float* qb     = (float*)wsAlloc((size_t)NNODE * QKDIM * 4);
  float* kb     = (float*)wsAlloc((size_t)NNODE * QKDIM * 4);
  float* vb     = (float*)wsAlloc((size_t)NNODE * QKDIM * 4);
  float* msg    = (float*)wsAlloc((size_t)NNODE * QKDIM * 4);
  float* edges  = (float*)wsAlloc((size_t)NEDGE * PAIR_CH * 4);
  float* ebias  = (float*)wsAlloc((size_t)NEDGE * HEADS * 4);
  float* logits = (float*)wsAlloc((size_t)NEDGE * HEADS * 4);
  float* aex    = (float*)wsAlloc((size_t)NEDGE * HEADS * 4);
  float* gl     = (float*)wsAlloc((size_t)NEDGE * HEADS * 4);
  unsigned* smax= (unsigned*)wsAlloc((size_t)NNODE * HEADS * 4);
  float* ssum   = (float*)wsAlloc((size_t)NNODE * HEADS * 4);
  int* srcA     = (int*)wsAlloc((size_t)NEDGE * 4);
  int* dstA     = (int*)wsAlloc((size_t)NEDGE * 4);
  float* tprop  = (float*)wsAlloc((size_t)CNUIJK * 24 * 4);
  float* pkern  = (float*)wsAlloc((size_t)CNIJKL * HEADS * 4);
  float* pex    = (float*)wsAlloc((size_t)CNIJKL * HEADS * 4);
  float* palpha = (float*)wsAlloc((size_t)CNIJKL * HEADS * 4);
  unsigned* pmax= (unsigned*)wsAlloc((size_t)CNIJK * HEADS * 4);
  float* psum   = (float*)wsAlloc((size_t)CNIJK * HEADS * 4);
  float* pouts  = (float*)wsAlloc((size_t)CNUIJK * 40 * 4);
  float* ptb    = (float*)wsAlloc((size_t)CNUIJK * 40 * 4);
  float* pmo    = (float*)wsAlloc((size_t)CNUIJK * PAIR_CH * 4);
  half_t* packb = (half_t*)wsAlloc((size_t)512 * 512 * sizeof(half_t));

  float* pair_s   = edges;
  float* prop_s   = edges + (size_t)CNUV * PAIR_CH;
  float* efa_s    = edges + (size_t)(CNUV + CNUIJK) * PAIR_CH;
  float* stereo_s = edges + (size_t)(CNUV + 2 * CNUIJK) * PAIR_CH;

  // ---- input projections ----
  run_gemm_t<23, 512, 512, 0, 0>(stream, ae,    in_atom_w,   in_atom_b,   nodes,    CNU,    packb);
  k_fill_f32<<<CDIV((long)CNIJK * NODE_CH, 256), 256, 0, stream>>>(nodes + (size_t)CNU * NODE_CH, 0.f, (long)CNIJK * NODE_CH);
  run_gemm_t< 8, 128, 128, 0, 0>(stream, ape,   in_pair_w,   in_pair_b,   pair_s,   CNUV,   packb);
  run_gemm_t<24,  24,  24, 1, 0>(stream, afpe,  in_prop_w1,  in_prop_b1,  tprop,    CNUIJK, packb);
  run_gemm_t<24, 128, 128, 0, 0>(stream, tprop, in_prop_w2,  in_prop_b2,  prop_s,   CNUIJK, packb);
  d2d4(stream, efa_s, prop_s, (long)CNUIJK * PAIR_CH);
  run_gemm_t<14, 128, 128, 0, 0>(stream, se,    in_stereo_w, in_stereo_b, stereo_s, CNIJKL, packb);
  k_build_sd<<<CDIV(NEDGE, 256), 256, 0, stream>>>(srcA, dstA, uv_u, uv_v, Uijk_u, Uijk_ijk, ijkl_ijk, ijkl_jkl);

  // ---- stacks ----
  for (int s = 0; s < N_STACKS; ++s) {
    const StackP& S = st[s];
    // -- transformer layer --
    run_ln(stream, nodes, S.ln1_s, S.ln1_b, hbuf, NNODE, NODE_CH);
    run_gemm_t<512, 256, 256, 0, 0>(stream, hbuf, S.wq, nullptr, qb, NNODE, packb);
    run_gemm_t<512, 256, 256, 0, 0>(stream, hbuf, S.wk, nullptr, kb, NNODE, packb);
    run_gemm_t<512, 256, 256, 0, 0>(stream, hbuf, S.wv, nullptr, vb, NNODE, packb);
    run_gemm_t<128,   8,   8, 0, 0>(stream, edges, S.wb, nullptr, ebias, NEDGE, packb);
    k_logits<<<CDIV((long)NEDGE * HEADS, 256), 256, 0, stream>>>(qb, kb, srcA, dstA, ebias, logits, NEDGE);
    k_fill_u32<<<CDIV((long)NNODE * HEADS, 256), 256, 0, stream>>>(smax, NEG_INF_ENC, (long)NNODE * HEADS);
    k_fill_f32<<<CDIV((long)NNODE * HEADS, 256), 256, 0, stream>>>(ssum, 0.f, (long)NNODE * HEADS);
    k_seg_max   <<<CDIV((long)NEDGE * HEADS, 256), 256, 0, stream>>>(logits, dstA, smax, NEDGE);
    k_seg_expsum<<<CDIV((long)NEDGE * HEADS, 256), 256, 0, stream>>>(logits, dstA, smax, aex, ssum, NEDGE);
    k_fill_f32<<<CDIV((long)NNODE * QKDIM, 256), 256, 0, stream>>>(msg, 0.f, (long)NNODE * QKDIM);
    k_attn_msg<<<CDIV((long)NEDGE * HEADS, 256), 256, 0, stream>>>(aex, ssum, vb, srcA, dstA, msg, NEDGE);
    run_gemm_t<256, 512, 512, 0, 1>(stream, msg, S.wo, S.bo, nodes, NNODE, packb);
    run_ln(stream, nodes, S.ln2_s, S.ln2_b, hbuf, NNODE, NODE_CH);
    run_gemm_t<512, 512, 512, 1, 0>(stream, hbuf, S.mlp_w1, S.mlp_b1, mlpt,  NNODE, packb);
    run_gemm_t<512, 512, 512, 0, 1>(stream, mlpt, S.mlp_w2, S.mlp_b2, nodes, NNODE, packb);
    k_gelu_arr<<<CDIV((long)NEDGE * HEADS, 256), 256, 0, stream>>>(gl, logits, (long)NEDGE * HEADS);
    run_gemm_t<  8, 128, 128, 0, 1>(stream, gl, S.w_eu, S.b_eu, edges, NEDGE, packb);
    // -- path_conv (prop = 2*prop_tl + gelu_mlp(concat outs)) --
    run_gemm_t<128,   8,   8, 0, 0>(stream, stereo_s, S.w_tri, nullptr, pkern, CNIJKL, packb);
    k_fill_u32<<<CDIV((long)CNIJK * HEADS, 256), 256, 0, stream>>>(pmax, NEG_INF_ENC, (long)CNIJK * HEADS);
    k_fill_f32<<<CDIV((long)CNIJK * HEADS, 256), 256, 0, stream>>>(psum, 0.f, (long)CNIJK * HEADS);
    k_seg_max   <<<CDIV((long)CNIJKL * HEADS, 256), 256, 0, stream>>>(pkern, ijkl_jkl, pmax, CNIJKL);
    k_seg_expsum<<<CDIV((long)CNIJKL * HEADS, 256), 256, 0, stream>>>(pkern, ijkl_jkl, pmax, pex, psum, CNIJKL);
    k_seg_norm  <<<CDIV((long)CNIJKL * HEADS, 256), 256, 0, stream>>>(pex, psum, ijkl_jkl, palpha, CNIJKL);
    run_gemm_t<128,   8,  40, 0, 0>(stream, prop_s, S.w_val, nullptr, pouts, CNUIJK, packb);
    k_zero_path_cols<<<CDIV((long)CNUIJK * 32, 256), 256, 0, stream>>>(pouts);
    for (int it = 0; it < MAX_PI; ++it)
      k_path_iter<<<CDIV((long)CNUIJKL * HEADS, 256), 256, 0, stream>>>(pouts, palpha, Uijkl_ijkl, Uijkl_Uijk, Uijkl_ujkl, it);
    run_gemm_t< 40,  40,  40, 1, 0>(stream, pouts, S.pmlp_w1, S.pmlp_b1, ptb, CNUIJK, packb);
    run_gemm_t< 40, 128, 128, 0, 0>(stream, ptb,   S.pmlp_w2, S.pmlp_b2, pmo, CNUIJK, packb);
    k_prop_fin<<<CDIV((long)CNUIJK * PAIR_CH, 256), 256, 0, stream>>>(prop_s, pmo, (long)CNUIJK * PAIR_CH);
  }

  // ---- outputs: atom, trp, pair, prop, af (flat concat) ----
  float* o = (float*)d_out;
  d2d4(stream, o, nodes, (long)CNU * NODE_CH);
  o += (size_t)CNU * NODE_CH;
  k_gather_rows4<<<CDIV((long)CNI * (NODE_CH / 4), 256), 256, 0, stream>>>(
      reinterpret_cast<float4*>(o), reinterpret_cast<const float4*>(nodes), I_ijk, CNI, NODE_CH / 4, CNU);
  o += (size_t)CNI * NODE_CH;
  d2d4(stream, o, pair_s, (long)CNUV * PAIR_CH);
  o += (size_t)CNUV * PAIR_CH;
  d2d4(stream, o, prop_s, (long)CNUIJK * PAIR_CH);
  o += (size_t)CNUIJK * PAIR_CH;
  k_gather_rows4<<<CDIV((long)CNUI * (PAIR_CH / 4), 256), 256, 0, stream>>>(
      reinterpret_cast<float4*>(o), reinterpret_cast<const float4*>(prop_s), UI_Uijk, CNUI, PAIR_CH / 4, 0);
}
